// IICBLoss_73744588472624
// MI455X (gfx1250) — matrix operand order repaired and checked by reading, not compiled, not run on hardware
//
#include <hip/hip_runtime.h>

// Problem constants (match reference)
#define BDIM 8
#define CDIM 19
#define HDIM 512
#define WDIM 512
#define HW   (HDIM * WDIM)
#define CHW  (CDIM * HW)
#define NPIX (BDIM * HDIM * WDIM)

typedef __attribute__((ext_vector_type(16))) _Float16 v16h;
typedef __attribute__((ext_vector_type(8)))  float    v8f;

// Workspace float layout
#define WS_CNT    0      // 19 floats: per-class pixel count
#define WS_S      19     // 361 floats: segment sum of probs   [19][19]
#define WS_SQ     380    // 361 floats: segment sum of probs^2 [19][19]
#define WS_CE_NUM 741    // sum(pix_w * log(p_t + eps))
#define WS_CE_CNT 742    // sum(pix_w)
#define WS_UNC    743    // sum((1-conf)*log(1-conf+eps))
#define WS_DX     744    // sum |dprobs/dh|
#define WS_DY     745    // sum |dprobs/dw|
#define WS_DI     746    // dice intersection sum(argmax * t)
#define WS_DA     747    // sum(argmax)
#define WS_DT     748    // sum(t)
#define WS_TOT    749

// One wave (32 lanes) owns one full image row (b,h): 16 groups of 32 pixels.
// grid = 512 blocks x 256 threads = 4096 waves = BDIM*HDIM rows exactly.
__global__ __launch_bounds__(256) void iicb_main(const float* __restrict__ pred,
                                                 const int* __restrict__ tgt,
                                                 float* __restrict__ ws) {
    // per-wave staging: packed {f16 p, f16 p*p} per class; stride 21 -> bank-conflict-free
    __shared__ unsigned psh[8][32][21];
    __shared__ int      tsh[8][32];
    // block-level class accumulators
    __shared__ float s_sh[361];
    __shared__ float q_sh[361];
    __shared__ float c_sh[19];
    __shared__ float red[8][8];

    const int tid  = threadIdx.x;
    const int lane = tid & 31;
    const int wid  = tid >> 5;

    for (int i = tid; i < 361; i += 256) { s_sh[i] = 0.f; q_sh[i] = 0.f; }
    if (tid < 19) c_sh[tid] = 0.f;
    __syncthreads();

    const int row = blockIdx.x * 8 + wid;          // 0..4095
    const int b   = row >> 9;
    const int h   = row & 511;
    const int rowbase = b * CHW + h * WDIM;        // fits in int (<40M)

    // WMMA f32 accumulators: 2x2 tiles for s and sq
    v8f aS00 = {}, aS01 = {}, aS10 = {}, aS11 = {};
    v8f aQ00 = {}, aQ01 = {}, aQ10 = {}, aQ11 = {};

    float ce_num = 0.f, ce_cnt = 0.f, uncs = 0.f;
    float dxs = 0.f, dys = 0.f, di = 0.f, da = 0.f, dtm = 0.f;

    float carry[CDIM];                              // prev group's lane-31 probs
#pragma unroll
    for (int c = 0; c < CDIM; ++c) carry[c] = 0.f;

    const int  nclm   = lane & 15;                  // B/D column, D row offset base
    const bool lolane = lane < 16;
    const int  kbB    = lolane ? 0 : 16;            // B K-range base
    const int  mrow   = lane & 15;                  // A row within tile
    const int  cls1   = (16 + nclm > 18) ? 18 : (16 + nclm);  // clamped col-tile-1 class

    for (int it = 0; it < 16; ++it) {
        const int w    = it * 32 + lane;
        const int base = rowbase + w;

        // ---- self softmax (conf = 1/sumexp since max exp == 1) ----
        float x[CDIM];
        float m = -1e30f; int bi = 0;
#pragma unroll
        for (int c = 0; c < CDIM; ++c) {
            x[c] = pred[base + c * HW];
            if (x[c] > m) { m = x[c]; bi = c; }
        }
        float se = 0.f;
#pragma unroll
        for (int c = 0; c < CDIM; ++c) { x[c] = __expf(x[c] - m); se += x[c]; }
        const float inv = 1.0f / se;
        float p[CDIM];
#pragma unroll
        for (int c = 0; c < CDIM; ++c) p[c] = x[c] * inv;

        const int t = tgt[b * HW + h * WDIM + w];
        float pt = 0.f;
#pragma unroll
        for (int c = 0; c < CDIM; ++c) pt = (c == t) ? p[c] : pt;

        const float conf = inv;
        const float pw   = (conf > 0.5f) ? 1.f : 0.f;
        ce_num += pw * __logf(pt + 1e-6f);
        ce_cnt += pw;
        uncs   += (1.f - conf) * __logf(1.f - conf + 1e-6f);
        di  += (float)bi * (float)t;
        da  += (float)bi;
        dtm += (float)t;

        // ---- dy (W-direction TV) via in-wave shuffle + lane-31 carry ----
        {
            float dsum = 0.f;
#pragma unroll
            for (int c = 0; c < CDIM; ++c) {
                float up   = __shfl_up(p[c], 1, 32);
                float left = (lane == 0) ? carry[c] : up;
                dsum += fabsf(p[c] - left);
            }
            dys += (w > 0) ? dsum : 0.f;
#pragma unroll
            for (int c = 0; c < CDIM; ++c) carry[c] = __shfl(p[c], 31, 32);
        }

        // ---- dx (H-direction TV): recompute down-neighbor softmax (L2 hit) ----
        if (h < HDIM - 1) {                         // wave-uniform branch
            float y[CDIM];
            float mn = -1e30f;
#pragma unroll
            for (int c = 0; c < CDIM; ++c) {
                y[c] = pred[base + WDIM + c * HW];
                mn = fmaxf(mn, y[c]);
            }
            float sn = 0.f;
#pragma unroll
            for (int c = 0; c < CDIM; ++c) { y[c] = __expf(y[c] - mn); sn += y[c]; }
            const float invn = 1.0f / sn;
            float dsum = 0.f;
#pragma unroll
            for (int c = 0; c < CDIM; ++c) dsum += fabsf(y[c] * invn - p[c]);
            dxs += dsum;
        }

        // ---- per-class count ----
        atomicAdd(&c_sh[t], 1.0f);

        // ---- stage packed {f16 p, f16 p^2} + target for WMMA operand build ----
#pragma unroll
        for (int c = 0; c < CDIM; ++c) {
            _Float16 hp = (_Float16)p[c];
            _Float16 hq = (_Float16)(p[c] * p[c]);
            unsigned u = (unsigned)__builtin_bit_cast(unsigned short, hp)
                       | ((unsigned)__builtin_bit_cast(unsigned short, hq) << 16);
            psh[wid][lane][c] = u;
        }
        tsh[wid][lane] = t;
        __builtin_amdgcn_wave_barrier();            // same-wave DS ops are in-order

        // ---- A: one-hot 16x32 f16 tiles (row = class, K = pixel) ----
        int tk[16];
#pragma unroll
        for (int j = 0; j < 16; ++j) {
            int k = (lolane ? 0 : 16) + j;
            tk[j] = tsh[wid][k];
        }
        v16h A0, A1;
#pragma unroll
        for (int j = 0; j < 16; ++j) {
            A0[j] = (tk[j] == mrow)      ? (_Float16)1.0f : (_Float16)0.0f;
            A1[j] = (tk[j] == mrow + 16) ? (_Float16)1.0f : (_Float16)0.0f;
        }

        // ---- B: probs 32x16 f16 tiles (row = pixel K, col = class) ----
        v16h B0p, B0q, B1p, B1q;
#pragma unroll
        for (int j = 0; j < 16; ++j) {
            unsigned u0 = psh[wid][kbB + j][nclm];
            unsigned u1 = psh[wid][kbB + j][cls1];
            B0p[j] = __builtin_bit_cast(_Float16, (unsigned short)(u0 & 0xffffu));
            B0q[j] = __builtin_bit_cast(_Float16, (unsigned short)(u0 >> 16));
            B1p[j] = __builtin_bit_cast(_Float16, (unsigned short)(u1 & 0xffffu));
            B1q[j] = __builtin_bit_cast(_Float16, (unsigned short)(u1 >> 16));
        }
        __builtin_amdgcn_wave_barrier();

        // ---- 8x v_wmma_f32_16x16x32_f16: D += OneHot x Probs (and Probs^2) ----
        aS00 = __builtin_amdgcn_wmma_f32_16x16x32_f16(false, A0, false, B0p, (short)0, aS00, false, false);
        aS01 = __builtin_amdgcn_wmma_f32_16x16x32_f16(false, A0, false, B1p, (short)0, aS01, false, false);
        aS10 = __builtin_amdgcn_wmma_f32_16x16x32_f16(false, A1, false, B0p, (short)0, aS10, false, false);
        aS11 = __builtin_amdgcn_wmma_f32_16x16x32_f16(false, A1, false, B1p, (short)0, aS11, false, false);
        aQ00 = __builtin_amdgcn_wmma_f32_16x16x32_f16(false, A0, false, B0q, (short)0, aQ00, false, false);
        aQ01 = __builtin_amdgcn_wmma_f32_16x16x32_f16(false, A0, false, B1q, (short)0, aQ01, false, false);
        aQ10 = __builtin_amdgcn_wmma_f32_16x16x32_f16(false, A1, false, B0q, (short)0, aQ10, false, false);
        aQ11 = __builtin_amdgcn_wmma_f32_16x16x32_f16(false, A1, false, B1q, (short)0, aQ11, false, false);
    }

    // ---- merge per-wave WMMA tiles into block LDS (D: VGPR r, M=r+(lane<16?0:8), N=lane&15) ----
    const int mofs = lolane ? 0 : 8;
#pragma unroll
    for (int r = 0; r < 8; ++r) {
        const int mg0 = r + mofs;
        const int mg1 = 16 + r + mofs;
        const int ng0 = nclm;
        const int ng1 = 16 + nclm;
        atomicAdd(&s_sh[mg0 * 19 + ng0], aS00[r]);
        atomicAdd(&q_sh[mg0 * 19 + ng0], aQ00[r]);
        if (ng1 < 19) {
            atomicAdd(&s_sh[mg0 * 19 + ng1], aS01[r]);
            atomicAdd(&q_sh[mg0 * 19 + ng1], aQ01[r]);
        }
        if (mg1 < 19) {
            atomicAdd(&s_sh[mg1 * 19 + ng0], aS10[r]);
            atomicAdd(&q_sh[mg1 * 19 + ng0], aQ10[r]);
            if (ng1 < 19) {
                atomicAdd(&s_sh[mg1 * 19 + ng1], aS11[r]);
                atomicAdd(&q_sh[mg1 * 19 + ng1], aQ11[r]);
            }
        }
    }

    // ---- scalar wave reduction -> block reduction -> global atomics ----
    float vals[8] = { ce_num, ce_cnt, uncs, dxs, dys, di, da, dtm };
#pragma unroll
    for (int i = 0; i < 8; ++i) {
        float v = vals[i];
        for (int off = 16; off > 0; off >>= 1) v += __shfl_xor(v, off, 32);
        if (lane == 0) red[wid][i] = v;
    }
    __syncthreads();
    if (tid < 8) {
        float v = 0.f;
        for (int wv = 0; wv < 8; ++wv) v += red[wv][tid];
        atomicAdd(&ws[WS_CE_NUM + tid], v);
    }
    for (int i = tid; i < 361; i += 256) {
        atomicAdd(&ws[WS_S  + i], s_sh[i]);
        atomicAdd(&ws[WS_SQ + i], q_sh[i]);
    }
    if (tid < 19) atomicAdd(&ws[WS_CNT + tid], c_sh[tid]);
}

// Tiny finalize: 19x19 class statistics + scalar combination.
__global__ __launch_bounds__(256) void iicb_final(const float* __restrict__ ws,
                                                  float* __restrict__ out) {
    __shared__ float mean[361];
    __shared__ float nmean[361];
    __shared__ float cnt[19];
    __shared__ float intra_c[19];
    __shared__ float norms[19];
    __shared__ float acc[3];   // 0: intra sum, 1: pair sum, 2: #present
    const int tid = threadIdx.x;
    if (tid < 3)  acc[tid] = 0.f;
    if (tid < 19) cnt[tid] = ws[WS_CNT + tid];
    __syncthreads();
    if (tid < 361) {
        const int i = tid / 19;
        mean[tid] = ws[WS_S + tid] / fmaxf(cnt[i], 1.0f);
    }
    __syncthreads();
    if (tid < 19) {
        float nrm2 = 0.f, intra = 0.f;
        for (int j = 0; j < 19; ++j) {
            const float mn = mean[tid * 19 + j];
            nrm2  += mn * mn;
            intra += ws[WS_SQ + tid * 19 + j] - cnt[tid] * mn * mn;
        }
        const float ic = intra / (fmaxf(cnt[tid], 1.f) * (float)CDIM);
        intra_c[tid] = (cnt[tid] > 1.f) ? ic : 0.f;
        norms[tid]   = sqrtf(nrm2);
        atomicAdd(&acc[0], intra_c[tid]);
        if (cnt[tid] > 0.f) atomicAdd(&acc[2], 1.f);
    }
    __syncthreads();
    if (tid < 361) {
        const int i = tid / 19;
        nmean[tid] = mean[tid] / fmaxf(norms[i], 1e-8f);
    }
    __syncthreads();
    if (tid < 361) {
        const int i = tid / 19, j = tid % 19;
        if (j > i && cnt[i] > 0.f && cnt[j] > 0.f) {
            float d = 0.f;
            for (int q = 0; q < 19; ++q) d += nmean[i * 19 + q] * nmean[j * 19 + q];
            atomicAdd(&acc[1], d);
        }
    }
    __syncthreads();
    if (tid == 0) {
        const float ce     = -ws[WS_CE_NUM] / ((float)CDIM * ws[WS_CE_CNT] + 1e-6f);
        const float unc    = -ws[WS_UNC] / (float)NPIX;
        const float smooth = ws[WS_DX] / ((float)BDIM * CDIM * (HDIM - 1) * WDIM)
                           + ws[WS_DY] / ((float)BDIM * CDIM * HDIM * (WDIM - 1));
        const float intra  = acc[0] / (float)CDIM;
        const float k      = acc[2];
        const float npairs = k * (k - 1.f) * 0.5f;
        const float inter  = (k > 1.f) ? (acc[1] / fmaxf(npairs, 1.f)) : 0.f;
        const float dice   = 1.f - (2.f * ws[WS_DI] + 1e-6f) / (ws[WS_DA] + ws[WS_DT] + 1e-6f);
        out[0] = 0.6f * ce + 0.3f * unc + 0.1f * smooth
               + 0.1f * (intra + (1.f - inter)) + 0.1f * dice;
    }
}

extern "C" void kernel_launch(void* const* d_in, const int* in_sizes, int n_in,
                              void* d_out, int out_size, void* d_ws, size_t ws_size,
                              hipStream_t stream) {
    (void)in_sizes; (void)n_in; (void)out_size; (void)ws_size;
    const float* pred = (const float*)d_in[0];
    const int*   tgt  = (const int*)d_in[1];
    float*       ws   = (float*)d_ws;

    hipMemsetAsync(d_ws, 0, WS_TOT * sizeof(float), stream);
    iicb_main <<<512, 256, 0, stream>>>(pred, tgt, ws);
    iicb_final<<<1,   256, 0, stream>>>(ws, (float*)d_out);
}
